// CausalSelfAttention_57432302682352
// MI455X (gfx1250) — matrix operand order repaired and checked by reading, not compiled, not run on hardware
//
#include <hip/hip_runtime.h>

// ---------------------------------------------------------------------------
// CDNA5 (gfx1250) causal self-attention: bf16 WMMA GEMMs + flash attention,
// with async global->LDS copies (ASYNCcnt) and double-buffered GEMM tiles.
// ---------------------------------------------------------------------------

typedef __bf16 bf16_t;
typedef __attribute__((ext_vector_type(16))) __bf16 v16bf;
typedef __attribute__((ext_vector_type(8)))  __bf16 v8bf;
typedef __attribute__((ext_vector_type(8)))  float  v8f;
typedef __attribute__((ext_vector_type(4)))  int    v4i;

#define N_HEAD 12
#define SEQ_T  2048
#define DIMC   768
#define HEADD  64

// ---------------- async global -> LDS (gfx1250) ----------------------------
#if __has_builtin(__builtin_amdgcn_global_load_async_to_lds_b128)
#define ASYNC_LDS 1
#else
#define ASYNC_LDS 0
#endif

__device__ __forceinline__ void copy16_to_lds(bf16_t* l, const bf16_t* g) {
#if ASYNC_LDS
  __builtin_amdgcn_global_load_async_to_lds_b128((v4i*)g, (v4i*)l, 0, 0);
#else
  *(v8bf*)l = *(const v8bf*)g;
#endif
}

template <int N>
__device__ __forceinline__ void wait_async() {
#if ASYNC_LDS
#if __has_builtin(__builtin_amdgcn_s_wait_asynccnt)
  __builtin_amdgcn_s_wait_asynccnt((short)N);
#else
  asm volatile("s_wait_asynccnt %0" :: "n"(N) : "memory");
#endif
#endif
}

// ---------------- WMMA helpers ---------------------------------------------
__device__ __forceinline__ v8f wmma_bf16(v16bf a, v16bf b, v8f c) {
  return __builtin_amdgcn_wmma_f32_16x16x32_bf16(
      /*neg_a=*/false, a, /*neg_b=*/false, b,
      /*c_mod=*/(short)0, c, /*reuse_a=*/false, /*reuse_b=*/false);
}

// A fragment (16x32 bf16). lane<16: row=lane, K {0..7,16..23}; lane>=16: K+8.
__device__ __forceinline__ v16bf load_frag_a(const bf16_t* base, int ld) {
  int lane = threadIdx.x & 31;
  int row  = lane & 15;
  int hi   = lane >> 4;
  const bf16_t* p = base + row * ld + hi * 8;
  v8bf c0 = *(const v8bf*)(p);
  v8bf c1 = *(const v8bf*)(p + 16);
  return __builtin_shufflevector(c0, c1, 0,1,2,3,4,5,6,7,8,9,10,11,12,13,14,15);
}

// B fragment (32x16 bf16) from B^T stored row-major [N x K].
// lane<16: col=lane, K {0..7,16..23}; lane>=16: K+8.
__device__ __forceinline__ v16bf load_frag_b(const bf16_t* base, int ld) {
  int lane = threadIdx.x & 31;
  int col  = lane & 15;
  int hi   = lane >> 4;
  const bf16_t* p = base + col * ld + hi * 8;
  v8bf c0 = *(const v8bf*)(p);
  v8bf c1 = *(const v8bf*)(p + 16);
  return __builtin_shufflevector(c0, c1, 0,1,2,3,4,5,6,7,8,9,10,11,12,13,14,15);
}

__device__ __forceinline__ v8f zero8() {
  v8f z = {0.f, 0.f, 0.f, 0.f, 0.f, 0.f, 0.f, 0.f};
  return z;
}

// ---------------------------------------------------------------------------
// fp32 -> bf16 convert
// ---------------------------------------------------------------------------
__global__ __launch_bounds__(256) void cvt_f32_bf16(const float* __restrict__ src,
                                                    bf16_t* __restrict__ dst, int n) {
  int i = blockIdx.x * 256 + threadIdx.x;
  if (i < n) dst[i] = (bf16_t)src[i];
}

// ---------------------------------------------------------------------------
// Tiled bf16 GEMM: out[m,n] = sum_k A[m,k] * Bt[n,k]  (A, Bt row-major, ld=K)
// BM=128, BN=128, BK=32, 8 waves -> each 32x64 (2x4 wmma tiles).
// Double-buffered LDS stages fed by async global->LDS copies.
// mode 0: QKV epilogue (q[B,H,T,D], k[B,H,T,D], vT[B,H,D,T], bf16)
// mode 1: f32 epilogue into outF (ld = 768)
// ---------------------------------------------------------------------------
#define BM 128
#define BN 128
#define BK 32
#define LDT 40   // padded LDS pitch (bf16); 80B rows, 16B aligned

__global__ __launch_bounds__(256) void gemm_bf16(
    const bf16_t* __restrict__ A, const bf16_t* __restrict__ Bt, int K, int mode,
    bf16_t* __restrict__ outQ, bf16_t* __restrict__ outK, bf16_t* __restrict__ outVt,
    float* __restrict__ outF)
{
  __shared__ __align__(16) bf16_t As[2][BM * LDT];
  __shared__ __align__(16) bf16_t Bs[2][BN * LDT];

  const int tid  = threadIdx.x;
  const int lane = tid & 31;
  const int wave = tid >> 5;
  const int wm   = wave & 3;   // 0..3 -> 32-row strip
  const int wn   = wave >> 2;  // 0..1 -> 64-col strip
  const int bm0  = blockIdx.x * BM;
  const int bn0  = blockIdx.y * BN;

  v8f acc[2][4];
#pragma unroll
  for (int mt = 0; mt < 2; ++mt)
#pragma unroll
    for (int nt = 0; nt < 4; ++nt) acc[mt][nt] = zero8();

  // One stage = 128x32 A tile + 128x32 B tile = 4 async b128 ops per thread.
  auto issue_stage = [&](int k0, int buf) {
#pragma unroll
    for (int i = 0; i < 2; ++i) {
      int c   = tid + i * 256;
      int row = c >> 2;
      int off = (c & 3) * 8;
      copy16_to_lds(&As[buf][row * LDT + off],
                    &A [(size_t)(bm0 + row) * K + k0 + off]);
      copy16_to_lds(&Bs[buf][row * LDT + off],
                    &Bt[(size_t)(bn0 + row) * K + k0 + off]);
    }
  };

  const int nsteps = K / BK;
  issue_stage(0, 0);

  for (int s = 0; s < nsteps; ++s) {
    const int buf = s & 1;
    if (s + 1 < nsteps) {
      issue_stage((s + 1) * BK, buf ^ 1);
      wait_async<4>();               // stage s landed (4 newer ops in flight)
    } else {
      wait_async<0>();
    }
    __syncthreads();

    v16bf af[2], bfm[4];
#pragma unroll
    for (int mt = 0; mt < 2; ++mt)
      af[mt] = load_frag_a(&As[buf][(wm * 32 + mt * 16) * LDT], LDT);
#pragma unroll
    for (int nt = 0; nt < 4; ++nt)
      bfm[nt] = load_frag_b(&Bs[buf][(wn * 64 + nt * 16) * LDT], LDT);

#pragma unroll
    for (int mt = 0; mt < 2; ++mt)
#pragma unroll
      for (int nt = 0; nt < 4; ++nt)
        acc[mt][nt] = wmma_bf16(af[mt], bfm[nt], acc[mt][nt]);

    __syncthreads();
  }

  // Epilogue. C/D layout: element (m = r + 8*hi, n = lane&15) in acc[..][r].
  const int hi  = lane >> 4;
  const int nlo = lane & 15;

  if (mode == 0) {
    const int b  = bm0 >> 11;                        // batch (2048 rows each)
    const int t0 = (bm0 & (SEQ_T - 1)) + wm * 32 + 8 * hi;
#pragma unroll
    for (int mt = 0; mt < 2; ++mt) {
      const int t = t0 + mt * 16;
#pragma unroll
      for (int nt = 0; nt < 4; ++nt) {
        const int n0    = bn0 + wn * 64 + nt * 16;   // uniform per tile
        const int which = n0 / DIMC;                 // 0=q, 1=k, 2=v
        const int c0    = n0 - which * DIMC;
        const int h     = c0 >> 6;
        const int d     = (c0 & (HEADD - 1)) + nlo;
        const size_t bhh = (size_t)(b * N_HEAD + h);
        if (which == 2) {
          bf16_t* dst = outVt + (bhh * HEADD + d) * SEQ_T + t;   // [B,H,D,T]
#pragma unroll
          for (int r = 0; r < 8; ++r) dst[r] = (bf16_t)acc[mt][nt][r];
        } else {
          bf16_t* dst = (which == 0 ? outQ : outK) +
                        (bhh * SEQ_T + t) * HEADD + d;           // [B,H,T,D]
#pragma unroll
          for (int r = 0; r < 8; ++r)
            dst[(size_t)r * HEADD] = (bf16_t)acc[mt][nt][r];
        }
      }
    }
  } else {
    const int m0 = bm0 + wm * 32 + 8 * hi;
#pragma unroll
    for (int mt = 0; mt < 2; ++mt)
#pragma unroll
      for (int nt = 0; nt < 4; ++nt) {
        float* dst = outF + (size_t)(m0 + mt * 16) * DIMC +
                     bn0 + wn * 64 + nt * 16 + nlo;
#pragma unroll
        for (int r = 0; r < 8; ++r) dst[(size_t)r * DIMC] = acc[mt][nt][r];
      }
  }
}

// ---------------------------------------------------------------------------
// Flash attention: grid (T/128, B*H), 8 waves/WG, wave = 16 query rows.
// ---------------------------------------------------------------------------
#define FA_LDK 72   // K-tile pitch (64+8), 144B rows
#define FA_LDV 40   // Vt-tile pitch (32+8), 80B rows
#define FA_LDP 40   // P-tile pitch (32+8), 80B rows

__global__ __launch_bounds__(256) void flash_attn(
    const bf16_t* __restrict__ Q, const bf16_t* __restrict__ Km,
    const bf16_t* __restrict__ Vt, bf16_t* __restrict__ Ya)
{
  __shared__ __align__(16) bf16_t Kls[32 * FA_LDK];       // keys x d
  __shared__ __align__(16) bf16_t Vls[HEADD * FA_LDV];    // d x keys
  __shared__ __align__(16) bf16_t Pls[8][16 * FA_LDP];    // per-wave P

  const int tid  = threadIdx.x;
  const int lane = tid & 31;
  const int wave = tid >> 5;
  const int hi   = lane >> 4;
  const int nlo  = lane & 15;

  const int qblk = blockIdx.x;            // 0..15
  const int bh   = blockIdx.y;            // 0..23  (b*12 + h)
  const size_t base = (size_t)bh * SEQ_T * HEADD;
  const bf16_t* q  = Q  + base;           // [T][64]
  const bf16_t* kk = Km + base;           // [T][64]
  const bf16_t* vt = Vt + base;           // [64][T]

  const int qbase_wg = qblk * 128;
  const int q0       = qbase_wg + wave * 16;

  v16bf qf[2];
  qf[0] = load_frag_a(q + (size_t)q0 * HEADD +  0, HEADD);
  qf[1] = load_frag_a(q + (size_t)q0 * HEADD + 32, HEADD);

  v8f oacc[4];
#pragma unroll
  for (int nt = 0; nt < 4; ++nt) oacc[nt] = zero8();
  float mstat[8], lstat[8];
#pragma unroll
  for (int r = 0; r < 8; ++r) { mstat[r] = -3.0e38f; lstat[r] = 0.0f; }

  const float scale = 0.125f;             // 1/sqrt(64)
  const int nkb = (qbase_wg >> 5) + 4;    // key blocks up to the diagonal

  for (int kb = 0; kb < nkb; ++kb) {
    const int kbase = kb * 32;

    // K block 32x64 and Vt block 64x32 -> LDS (async, 2 ops/thread).
    {
      int row = tid >> 3, off = (tid & 7) * 8;
      copy16_to_lds(&Kls[row * FA_LDK + off],
                    &kk[(size_t)(kbase + row) * HEADD + off]);
    }
    {
      int row = tid >> 2, off = (tid & 3) * 8;
      copy16_to_lds(&Vls[row * FA_LDV + off],
                    &vt[(size_t)row * SEQ_T + kbase + off]);
    }
    wait_async<0>();
    __syncthreads();

    // S = Q K^T : two 16x16 key tiles, K-dim = 64 (2 wmma k-steps).
    v8f s[2];
    s[0] = zero8(); s[1] = zero8();
#pragma unroll
    for (int ks = 0; ks < 2; ++ks) {
      v16bf kf0 = load_frag_b(&Kls[ 0 * FA_LDK + ks * 32], FA_LDK);
      v16bf kf1 = load_frag_b(&Kls[16 * FA_LDK + ks * 32], FA_LDK);
      s[0] = wmma_bf16(qf[ks], kf0, s[0]);
      s[1] = wmma_bf16(qf[ks], kf1, s[1]);
    }

    // Scale + causal mask + row max (rows uniform across 16-lane halves).
    float rowmax[8];
#pragma unroll
    for (int r = 0; r < 8; ++r) {
      int qg = q0 + r + 8 * hi;
      float s0 = s[0][r] * scale;
      float s1 = s[1][r] * scale;
      if (kbase + nlo      > qg) s0 = -3.0e38f;
      if (kbase + 16 + nlo > qg) s1 = -3.0e38f;
      s[0][r] = s0; s[1][r] = s1;
      float mx = fmaxf(s0, s1);
      mx = fmaxf(mx, __shfl_xor(mx, 1, 32));
      mx = fmaxf(mx, __shfl_xor(mx, 2, 32));
      mx = fmaxf(mx, __shfl_xor(mx, 4, 32));
      mx = fmaxf(mx, __shfl_xor(mx, 8, 32));
      rowmax[r] = mx;
    }

    // Online softmax update.
#pragma unroll
    for (int r = 0; r < 8; ++r) {
      float mnew  = fmaxf(mstat[r], rowmax[r]);
      float alpha = __expf(mstat[r] - mnew);
      float p0 = __expf(s[0][r] - mnew);
      float p1 = __expf(s[1][r] - mnew);
      s[0][r] = p0; s[1][r] = p1;
      float rs = p0 + p1;
      rs += __shfl_xor(rs, 1, 32);
      rs += __shfl_xor(rs, 2, 32);
      rs += __shfl_xor(rs, 4, 32);
      rs += __shfl_xor(rs, 8, 32);
      lstat[r] = lstat[r] * alpha + rs;
      mstat[r] = mnew;
#pragma unroll
      for (int nt = 0; nt < 4; ++nt) oacc[nt][r] *= alpha;
    }

    // P (C/D layout) -> LDS -> A layout (wave-private buffer).
#pragma unroll
    for (int r = 0; r < 8; ++r) {
      int m = r + 8 * hi;
      Pls[wave][m * FA_LDP + nlo]      = (bf16_t)s[0][r];
      Pls[wave][m * FA_LDP + 16 + nlo] = (bf16_t)s[1][r];
    }
    v16bf pf = load_frag_a(Pls[wave], FA_LDP);

    // O += P V (B^T rows = d, from Vt tile).
#pragma unroll
    for (int nt = 0; nt < 4; ++nt) {
      v16bf vf = load_frag_b(&Vls[nt * 16 * FA_LDV], FA_LDV);
      oacc[nt] = wmma_bf16(pf, vf, oacc[nt]);
    }
    __syncthreads();
  }

  // Normalize and store into [B, T, C] bf16 (C = h*64 + d).
  const int b = bh / N_HEAD;
  const int h = bh - b * N_HEAD;
#pragma unroll
  for (int r = 0; r < 8; ++r) {
    int qg = q0 + r + 8 * hi;
    float inv = 1.0f / lstat[r];
#pragma unroll
    for (int nt = 0; nt < 4; ++nt) {
      int d = nt * 16 + nlo;
      Ya[((size_t)(b * SEQ_T + qg)) * DIMC + h * HEADD + d] =
          (bf16_t)(oacc[nt][r] * inv);
    }
  }
}

// ---------------------------------------------------------------------------
// Host-side launch
// ---------------------------------------------------------------------------
extern "C" void kernel_launch(void* const* d_in, const int* in_sizes, int n_in,
                              void* d_out, int out_size, void* d_ws, size_t ws_size,
                              hipStream_t stream) {
  (void)in_sizes; (void)n_in; (void)out_size; (void)ws_size;

  const float* x  = (const float*)d_in[0];   // [2, 2048, 768]
  const float* Wa = (const float*)d_in[1];   // [2304, 768]
  const float* Wp = (const float*)d_in[2];   // [768, 768]
  float* out = (float*)d_out;                // [2, 2048, 768]

  const size_t nX  = (size_t)2 * SEQ_T * DIMC;
  const size_t nWa = (size_t)3 * DIMC * DIMC;
  const size_t nWp = (size_t)DIMC * DIMC;
  const size_t nH  = (size_t)2 * N_HEAD * SEQ_T * HEADD;

  char* ws = (char*)d_ws;
  size_t off = 0;
  auto take = [&](size_t elems) {
    bf16_t* p = (bf16_t*)(ws + off);
    off = (off + elems * sizeof(bf16_t) + 255) & ~(size_t)255;
    return p;
  };
  bf16_t* xb  = take(nX);
  bf16_t* wab = take(nWa);
  bf16_t* wpb = take(nWp);
  bf16_t* qb  = take(nH);
  bf16_t* kb  = take(nH);
  bf16_t* vtb = take(nH);
  bf16_t* ya  = take(nX);

  cvt_f32_bf16<<<(int)((nX  + 255) / 256), 256, 0, stream>>>(x,  xb,  (int)nX);
  cvt_f32_bf16<<<(int)((nWa + 255) / 256), 256, 0, stream>>>(Wa, wab, (int)nWa);
  cvt_f32_bf16<<<(int)((nWp + 255) / 256), 256, 0, stream>>>(Wp, wpb, (int)nWp);

  gemm_bf16<<<dim3(4096 / BM, 2304 / BN), 256, 0, stream>>>(
      xb, wab, DIMC, /*mode=*/0, qb, kb, vtb, nullptr);

  flash_attn<<<dim3(SEQ_T / 128, 2 * N_HEAD), 256, 0, stream>>>(qb, kb, vtb, ya);

  gemm_bf16<<<dim3(4096 / BM, 768 / BN), 256, 0, stream>>>(
      ya, wpb, DIMC, /*mode=*/1, nullptr, nullptr, nullptr, out);
}
